// MOELayer_30597347017387
// MI455X (gfx1250) — matrix operand order repaired and checked by reading, not compiled, not run on hardware
//
#include <hip/hip_runtime.h>

#define TOKENS      16384
#define IN_FEAT     512
#define OUT_FEAT    512
#define NUM_EXPERT  8
#define ROW_ALIGN   128
#define ROW_CAP     (TOKENS + NUM_EXPERT * ROW_ALIGN)   // 17408 padded sorted rows
#define NCHUNK      (IN_FEAT / 32)                      // 16 K-chunks
#define STRIDE_B    40                                  // padded halves per LDS B row

// ws layout:
//   ints:  [0..7] counts, [8..15] cursors, [16..24] offs(9), pad to 128,
//          [128 .. 128+ROW_CAP) row_token                (17536 ints = 70144 B)
//   bytes: A16 at 70144        (17408*512 f16 = 17,825,792 B)
//          W16 at 17,895,936   (8*512*512 f16 = 4 MB)
//          dump row at 22,090,240 (512 f32)
#define WS_ROWTOK_OFF   128
#define WS_A16_BYTES    70144
#define WS_W16_BYTES    17895936
#define WS_DUMP_BYTES   22090240

typedef _Float16 v16h __attribute__((ext_vector_type(16)));
typedef _Float16 v8h  __attribute__((ext_vector_type(8)));
typedef _Float16 v4h  __attribute__((ext_vector_type(4)));
typedef float    v8f  __attribute__((ext_vector_type(8)));
typedef int      v4i  __attribute__((ext_vector_type(4)));

#if defined(__has_builtin)
#if __has_builtin(__builtin_amdgcn_global_load_async_to_lds_b128)
#define ASYNC_OK 1
#endif
#endif

// Copy 16 f16 (two b128) global -> LDS for one thread's slice of a B chunk.
__device__ __forceinline__ void stage16(const _Float16* g, _Float16* l) {
#ifdef ASYNC_OK
    // builtin takes generic int4* pointers (per hipcc diagnostic); codegen
    // inserts the AS1/AS3 casts. Args: (global src, lds dst, imm off, imm cpol).
    __builtin_amdgcn_global_load_async_to_lds_b128((v4i*)g, (v4i*)l, 0, 0);
    __builtin_amdgcn_global_load_async_to_lds_b128((v4i*)(g + 8), (v4i*)(l + 8), 0, 0);
#else
    v8h t0 = *(const v8h*)g;
    v8h t1 = *(const v8h*)(g + 8);
    *(v8h*)l = t0;
    *(v8h*)(l + 8) = t1;
#endif
}

__device__ __forceinline__ void wait_async_2() {
#ifdef ASYNC_OK
#if __has_builtin(__builtin_amdgcn_s_wait_asynccnt)
    __builtin_amdgcn_s_wait_asynccnt(2);
#else
    asm volatile("s_wait_asynccnt 0x2" ::: "memory");
#endif
#endif
}

__global__ void moe_init(int* __restrict__ wsI) {
    int i = blockIdx.x * blockDim.x + threadIdx.x;
    if (i < 32) wsI[i] = 0;
    if (i < ROW_CAP) wsI[WS_ROWTOK_OFF + i] = -1;
}

__global__ void moe_count(const int* __restrict__ gate, int* __restrict__ wsI) {
    int t = blockIdx.x * blockDim.x + threadIdx.x;
    if (t < TOKENS) atomicAdd(&wsI[gate[t] & 7], 1);
}

__global__ void moe_offsets(int* __restrict__ wsI) {
    if (threadIdx.x == 0 && blockIdx.x == 0) {
        int off = 0;
        for (int e = 0; e < NUM_EXPERT; ++e) {
            wsI[16 + e] = off;
            off += (wsI[e] + ROW_ALIGN - 1) & ~(ROW_ALIGN - 1);
        }
        wsI[16 + NUM_EXPERT] = off;
    }
}

__global__ void moe_scatter(const int* __restrict__ gate, int* __restrict__ wsI) {
    int t = blockIdx.x * blockDim.x + threadIdx.x;
    if (t < TOKENS) {
        int e = gate[t] & 7;
        int pos = wsI[16 + e] + atomicAdd(&wsI[8 + e], 1);
        wsI[WS_ROWTOK_OFF + pos] = t;
    }
}

__global__ void moe_cvt_w(const float* __restrict__ w, _Float16* __restrict__ w16) {
    int i = (blockIdx.x * blockDim.x + threadIdx.x) * 4;
    float4 f = *(const float4*)(w + i);
    v4h h;
    h[0] = (_Float16)f.x; h[1] = (_Float16)f.y;
    h[2] = (_Float16)f.z; h[3] = (_Float16)f.w;
    *(v4h*)(w16 + i) = h;
}

__global__ void moe_gather_a(const float* __restrict__ inp, const int* __restrict__ wsI,
                             _Float16* __restrict__ a16) {
    const int r = blockIdx.x;
    const int c = threadIdx.x * 2;
    const int t = wsI[WS_ROWTOK_OFF + r];
    float2 f;
    if (t >= 0) f = *(const float2*)(inp + (size_t)t * IN_FEAT + c);
    else        f = make_float2(0.0f, 0.0f);
    _Float16* p = a16 + (size_t)r * IN_FEAT + c;
    p[0] = (_Float16)f.x; p[1] = (_Float16)f.y;
}

// Block: 256 thr = 8 waves; tile 128 rows x 128 cols (wave: 16 x 128, 8 accs).
// B chunk (32K x 128 cols, 8 KB) double-buffered in LDS via async copies and
// shared by all 8 waves; A fragments stream from global (f16, pre-gathered).
__global__ __launch_bounds__(256)
void moe_gemm(const _Float16* __restrict__ a16, const _Float16* __restrict__ w16,
              const int* __restrict__ wsI, float* __restrict__ out,
              float* __restrict__ dump) {
    __shared__ __align__(16) _Float16 sB[2][128 * STRIDE_B];   // 2 x 10 KB
    __shared__ int s_offs[9];

    const int tid  = threadIdx.x;
    const int lane = tid & 31;
    const int wave = tid >> 5;
    const int rowbase = blockIdx.x * 128;
    const int colbase = blockIdx.y * 128;

    if (tid < 9) s_offs[tid] = wsI[16 + tid];
    __syncthreads();
    if (rowbase >= s_offs[8]) return;      // block-uniform

    int e = 0;
#pragma unroll
    for (int i = 1; i < NUM_EXPERT; ++i)
        if (rowbase >= s_offs[i]) e = i;

    const int h    = lane >> 4;
    const int mrow = lane & 15;
    const int ncol = lane & 15;
    const int wrow = rowbase + 16 * wave;

    const _Float16* aptr = a16 + (size_t)(wrow + mrow) * IN_FEAT;

    // Staging: thread moves 16 halves of one column per chunk.
    const int scol  = tid >> 1;            // 0..127 (local output column)
    const int shalf = (tid & 1) * 16;      // first/second 16 halves of the chunk
    const _Float16* gB = w16 + (size_t)(e * OUT_FEAT + colbase + scol) * IN_FEAT + shalf;
    _Float16* lB[2] = { &sB[0][scol * STRIDE_B + shalf], &sB[1][scol * STRIDE_B + shalf] };

    v8f acc[8];
#pragma unroll
    for (int j = 0; j < 8; ++j)
#pragma unroll
        for (int i = 0; i < 8; ++i) acc[j][i] = 0.0f;

    stage16(gB, lB[0]);                    // chunk 0 -> buf0
#ifndef ASYNC_OK
    __syncthreads();
#endif

    for (int c = 0; c < NCHUNK; ++c) {
        const int p  = c & 1;
        const int k0 = c * 32;
        const int kn = (c + 1 < NCHUNK) ? (k0 + 32) : 0;   // dummy re-issue on last
        stage16(gB + kn, lB[1 - p]);
#ifdef ASYNC_OK
        wait_async_2();                    // own chunk-c copies landed
        __syncthreads();                   // everyone's chunk c visible
#endif
        // A fragment (ISA 7.12.2): K = k0+8h+0..7 and k0+16+8h+0..7
        v8h a_lo = *(const v8h*)(aptr + k0 + 8 * h);
        v8h a_hi = *(const v8h*)(aptr + k0 + 16 + 8 * h);
        v16h a;
#pragma unroll
        for (int i = 0; i < 8; ++i) { a[i] = a_lo[i]; a[8 + i] = a_hi[i]; }

        // B fragments from LDS buf p: col-major rows of STRIDE_B halves
        const _Float16* bp = &sB[p][ncol * STRIDE_B + 8 * h];
        v16h b[8];
#pragma unroll
        for (int j = 0; j < 8; ++j) {
            v8h b0 = *(const v8h*)(bp + j * 16 * STRIDE_B);
            v8h b1 = *(const v8h*)(bp + j * 16 * STRIDE_B + 16);
#pragma unroll
            for (int i = 0; i < 8; ++i) { b[j][i] = b0[i]; b[j][8 + i] = b1[i]; }
        }
        asm volatile("" ::: "memory");     // keep ds loads grouped ahead of WMMAs
#pragma unroll
        for (int j = 0; j < 8; ++j)
            acc[j] = __builtin_amdgcn_wmma_f32_16x16x32_f16(
                false, a, false, b[j], (short)0, acc[j], false, false);
        __syncthreads();                   // all reads of buf p done -> reusable
    }

    // Branchless scatter; padding rows dump into ws.
    const int* row_token = wsI + WS_ROWTOK_OFF;
    float* obase[8];
#pragma unroll
    for (int r = 0; r < 8; ++r) {
        int t = row_token[wrow + 8 * h + r];
        obase[r] = (t >= 0) ? (out + (size_t)t * OUT_FEAT) : dump;
    }
#pragma unroll
    for (int j = 0; j < 8; ++j) {
        int o = colbase + 16 * j + ncol;
#pragma unroll
        for (int r = 0; r < 8; ++r)
            obase[r][o] = acc[j][r];
    }
}

extern "C" void kernel_launch(void* const* d_in, const int* in_sizes, int n_in,
                              void* d_out, int out_size, void* d_ws, size_t ws_size,
                              hipStream_t stream) {
    const float* inp    = (const float*)d_in[0];
    const int*   gate   = (const int*)d_in[1];
    const float* weight = (const float*)d_in[2];
    float*       out    = (float*)d_out;

    int*       wsI  = (int*)d_ws;
    _Float16*  a16  = (_Float16*)((char*)d_ws + WS_A16_BYTES);
    _Float16*  w16  = (_Float16*)((char*)d_ws + WS_W16_BYTES);
    float*     dump = (float*)((char*)d_ws + WS_DUMP_BYTES);

    moe_init    <<<(ROW_CAP + 255) / 256, 256, 0, stream>>>(wsI);
    moe_count   <<<TOKENS / 256,          256, 0, stream>>>(gate, wsI);
    moe_offsets <<<1,                      32, 0, stream>>>(wsI);
    moe_scatter <<<TOKENS / 256,          256, 0, stream>>>(gate, wsI);
    moe_cvt_w   <<<(NUM_EXPERT * OUT_FEAT * IN_FEAT / 4) / 256, 256, 0, stream>>>(weight, w16);
    moe_gather_a<<<ROW_CAP,               256, 0, stream>>>(inp, wsI, a16);

    dim3 ggrid(ROW_CAP / 128, OUT_FEAT / 128);
    moe_gemm    <<<ggrid, 256, 0, stream>>>(a16, w16, wsI, out, dump);
}
